// ECPoseTransformer_19988777796198
// MI455X (gfx1250) — compile-verified, matrix operand order checked by reading, not run on hardware
//
#include <hip/hip_runtime.h>
#include <hip/hip_bf16.h>
#include <math.h>

// ---------------------------------------------------------------------------
// MSDeformAttn pose forward for MI455X (gfx1250, wave32, WMMA)
//
// Static shapes: bs=4, nq=300, nkpt=18 -> lq=5400, d_model=256
// H=8 heads, L=3 levels (80x80, 40x40, 20x20), P=4 points, head_dim=32
// ---------------------------------------------------------------------------

#define BS 4
#define LQ 5400
#define DM 256
#define NH 8
#define NL 3
#define NP 4
#define HD 32
#define NROWS (BS * LQ)          // 21600
#define NCOLS (NL * NP * 2 * NH + NL * NP * NH) // 192 + 96 = 288
#define NOFF  (NH * NL * NP * 2) // 192
#define NATT  (NH * NL * NP)     // 96

typedef __attribute__((ext_vector_type(16))) _Float16 v16h;
typedef __attribute__((ext_vector_type(8)))  _Float16 v8h;
typedef __attribute__((ext_vector_type(8)))  float    v8f;

// ---------------------------------------------------------------------------
// 1) Convert query f32 -> f16 (row major [NROWS][256])
// ---------------------------------------------------------------------------
__global__ void cvt_query_f16(const float* __restrict__ q,
                              _Float16* __restrict__ qh, int n) {
    int i = blockIdx.x * 256 + threadIdx.x;
    if (i < n) qh[i] = (_Float16)q[i];
}

// ---------------------------------------------------------------------------
// 2) Build combined transposed weight WhT[n][k] (f16), n in [0,288), k in [0,256)
//    n <  192 : W_off[k][n]        (W_off is [256][192])
//    n >= 192 : W_attn[k][n-192]   (W_attn is [256][96])
// ---------------------------------------------------------------------------
__global__ void build_wT_f16(const float* __restrict__ W_off,
                             const float* __restrict__ W_attn,
                             _Float16* __restrict__ WhT) {
    int i = blockIdx.x * 256 + threadIdx.x;
    if (i >= NCOLS * DM) return;
    int n = i >> 8;        // column of combined projection
    int k = i & 255;       // reduction index
    float v = (n < NOFF) ? W_off[k * NOFF + n] : W_attn[k * NATT + (n - NOFF)];
    WhT[i] = (_Float16)v;
}

// ---------------------------------------------------------------------------
// 3) Value transpose: in[bh][c][S] -> out[bh][s][c]  (bh=32, c=32)
//    Classic 32x32 LDS tile transpose; makes per-sample channel reads
//    contiguous (128B per corner per wave).
// ---------------------------------------------------------------------------
__global__ __launch_bounds__(256) void transpose_vals(
    const float* __restrict__ in, float* __restrict__ out, int S) {
    __shared__ float tile[32][33];
    int bh = blockIdx.y;
    int s0 = blockIdx.x * 32;
    int tx = threadIdx.x;  // 0..31
    int ty = threadIdx.y;  // 0..7
    const float* ip = in + (size_t)bh * 32 * S;
    float*       op = out + (size_t)bh * S * 32;
#pragma unroll
    for (int i = 0; i < 32; i += 8) {
        int c = ty + i;
        int s = s0 + tx;
        tile[c][tx] = (s < S) ? ip[(size_t)c * S + s] : 0.0f;
    }
    __syncthreads();
#pragma unroll
    for (int i = 0; i < 32; i += 8) {
        int s = s0 + ty + i;
        if (s < S) op[(size_t)s * 32 + tx] = tile[tx][ty + i];
    }
}

// ---------------------------------------------------------------------------
// 4) Projection GEMM via WMMA: P[NROWS][288] = Qh[NROWS][256] * W[256][288] + b
//    One wave owns one 16x16 output tile; K=256 in 8 steps of 32.
//    A fragment (16-bit A 16x32 layout): lane m=lane&15, K-octet = (lane>>4)*8,
//      halves 0..7 -> K = kb+0..7, halves 8..15 -> K = 16+kb+0..7.
//    B fragment mirrors A with n=lane&15 over pre-transposed WhT[n][k].
//    C/D layout: VGPR r: lanes 0-15 -> M=r, lanes 16-31 -> M=r+8; N=lane&15.
// ---------------------------------------------------------------------------
__global__ __launch_bounds__(256) void proj_wmma(
    const _Float16* __restrict__ Qh,    // [NROWS][256]
    const _Float16* __restrict__ WhT,   // [288][256]
    const float* __restrict__ b_off,    // [192]
    const float* __restrict__ b_attn,   // [96]
    float* __restrict__ P)              // [NROWS][288]
{
    const int M_TILES = NROWS / 16;     // 1350
    int wave = threadIdx.x >> 5;
    int lane = threadIdx.x & 31;
    int mtile = blockIdx.y * 8 + wave;
    if (mtile >= M_TILES) return;       // wave-uniform exit; EXEC stays full
    int ntile = blockIdx.x;             // 0..17

    int mn  = lane & 15;
    int kb  = (lane >> 4) * 8;
    const _Float16* arow = Qh  + (size_t)(mtile * 16 + mn) * DM + kb;
    const _Float16* brow = WhT + (size_t)(ntile * 16 + mn) * DM + kb;

    v8f c = {};
#pragma unroll
    for (int ks = 0; ks < DM; ks += 32) {
        v8h alo = *(const v8h*)(arow + ks);
        v8h ahi = *(const v8h*)(arow + ks + 16);
        v8h blo = *(const v8h*)(brow + ks);
        v8h bhi = *(const v8h*)(brow + ks + 16);
        v16h a = __builtin_shufflevector(alo, ahi, 0,1,2,3,4,5,6,7,8,9,10,11,12,13,14,15);
        v16h b = __builtin_shufflevector(blo, bhi, 0,1,2,3,4,5,6,7,8,9,10,11,12,13,14,15);
        c = __builtin_amdgcn_wmma_f32_16x16x32_f16(
                /*neg_a=*/false, a, /*neg_b=*/false, b,
                /*c_mod=*/(short)0, c, /*reuse_a=*/false, /*reuse_b=*/false);
    }

    int col  = ntile * 16 + (lane & 15);
    float bias = (col < NOFF) ? b_off[col] : b_attn[col - NOFF];
    int mbase = mtile * 16 + ((lane >> 4) << 3);
#pragma unroll
    for (int r = 0; r < 8; ++r) {
        P[(size_t)(mbase + r) * NCOLS + col] = c[r] + bias;
    }
}

// ---------------------------------------------------------------------------
// 5) Fused softmax + bilinear sampling + weighted sum.
//    One wave per (b, q, head); lane = channel (head_dim=32).
//    Each corner gather is one coalesced 128B wave load from value_t[bh][s][c].
// ---------------------------------------------------------------------------
__global__ __launch_bounds__(256) void msda_sample(
    const float* __restrict__ ref_points,  // [4][300][3][18][2]
    const float* __restrict__ P,           // [NROWS][288]
    const float* __restrict__ vt0,         // [32][6400][32]
    const float* __restrict__ vt1,         // [32][1600][32]
    const float* __restrict__ vt2,         // [32][400][32]
    float* __restrict__ out)               // [NROWS][256]
{
    int lane = threadIdx.x & 31;
    int h    = threadIdx.x >> 5;           // head 0..7
    int bq   = blockIdx.x;                 // 0..21599
    int b    = bq / LQ;
    int q    = bq % LQ;
    int qi   = q / 18;
    int kp   = q % 18;

    const float* prow = P + (size_t)bq * NCOLS;

    // softmax over the 12 (level,point) logits of this head
    float lg[12];
    float mx = -1e30f;
#pragma unroll
    for (int j = 0; j < 12; ++j) {
        lg[j] = prow[NOFF + h * 12 + j];
        mx = fmaxf(mx, lg[j]);
    }
    float sum = 0.0f;
#pragma unroll
    for (int j = 0; j < 12; ++j) {
        lg[j] = __expf(lg[j] - mx);
        sum += lg[j];
    }
    float inv = 1.0f / sum;

    const float* offp = prow + h * (NL * NP * 2); // 24 floats per head
    const int   HW_W[3] = {80, 40, 20};
    const int   HW_H[3] = {80, 40, 20};
    const int   HW_S[3] = {6400, 1600, 400};
    const float* vts[3] = {vt0, vt1, vt2};
    int bh = b * NH + h;

    float acc = 0.0f;
#pragma unroll
    for (int l = 0; l < NL; ++l) {
        size_t ridx = ((((size_t)(b * 300 + qi) * NL + l) * 18 + kp) * 2);
        float rfx = ref_points[ridx + 0];
        float rfy = ref_points[ridx + 1];
        int W = HW_W[l], H = HW_H[l];
        const float* vbase = vts[l] + (size_t)bh * HW_S[l] * HD;
#pragma unroll
        for (int p = 0; p < NP; ++p) {
            float ox = offp[l * 8 + p * 2 + 0];
            float oy = offp[l * 8 + p * 2 + 1];
            // loc = ref + off/normalizer; x = loc.x*W - 0.5 -> ref.x*W + ox - 0.5
            float x = rfx * (float)W + ox - 0.5f;
            float y = rfy * (float)H + oy - 0.5f;
            float x0f = floorf(x), y0f = floorf(y);
            int x0 = (int)x0f, y0 = (int)y0f;
            float lx = x - x0f, ly = y - y0f;
            float aw  = lg[l * 4 + p] * inv;
            float w00 = (1.0f - lx) * (1.0f - ly) * aw;
            float w10 = lx * (1.0f - ly) * aw;
            float w01 = (1.0f - lx) * ly * aw;
            float w11 = lx * ly * aw;

            int x1 = x0 + 1, y1 = y0 + 1;
            bool vx0 = (x0 >= 0) & (x0 < W);
            bool vx1 = (x1 >= 0) & (x1 < W);
            bool vy0 = (y0 >= 0) & (y0 < H);
            bool vy1 = (y1 >= 0) & (y1 < H);

            if (vy0) {
                const float* r0 = vbase + ((size_t)y0 * W) * HD;
                if (vx0) acc += w00 * r0[(size_t)x0 * HD + lane];
                if (vx1) acc += w10 * r0[(size_t)x1 * HD + lane];
            }
            if (vy1) {
                const float* r1 = vbase + ((size_t)y1 * W) * HD;
                if (vx0) acc += w01 * r1[(size_t)x0 * HD + lane];
                if (vx1) acc += w11 * r1[(size_t)x1 * HD + lane];
            }
        }
    }
    out[(size_t)bq * DM + h * HD + lane] = acc;
}

// ---------------------------------------------------------------------------
// Launcher
// ---------------------------------------------------------------------------
extern "C" void kernel_launch(void* const* d_in, const int* in_sizes, int n_in,
                              void* d_out, int out_size, void* d_ws, size_t ws_size,
                              hipStream_t stream) {
    const float* query  = (const float*)d_in[0];  // [4][5400][256]
    const float* refpts = (const float*)d_in[1];  // [4][300][3][18][2]
    const float* value0 = (const float*)d_in[2];  // [32][32][6400]
    const float* value1 = (const float*)d_in[3];  // [32][32][1600]
    const float* value2 = (const float*)d_in[4];  // [32][32][400]
    const float* W_off  = (const float*)d_in[5];
    const float* b_off  = (const float*)d_in[6];
    const float* W_attn = (const float*)d_in[7];
    const float* b_attn = (const float*)d_in[8];
    float* out = (float*)d_out;

    // Workspace layout (bytes, 256-aligned)
    char* ws = (char*)d_ws;
    size_t oP   = 0;                                   // P:   21600*288 f32
    size_t oQh  = oP  + (size_t)NROWS * NCOLS * 4;     // Qh:  21600*256 f16
    size_t oWhT = oQh + (size_t)NROWS * DM * 2;        // WhT:   288*256 f16
    size_t oV0  = oWhT + (size_t)NCOLS * DM * 2;       // vt0: 32*6400*32 f32
    size_t oV1  = oV0 + (size_t)32 * 6400 * 32 * 4;    // vt1: 32*1600*32 f32
    size_t oV2  = oV1 + (size_t)32 * 1600 * 32 * 4;    // vt2: 32* 400*32 f32

    float*    P    = (float*)(ws + oP);
    _Float16* Qh   = (_Float16*)(ws + oQh);
    _Float16* WhT  = (_Float16*)(ws + oWhT);
    float*    vt0  = (float*)(ws + oV0);
    float*    vt1  = (float*)(ws + oV1);
    float*    vt2  = (float*)(ws + oV2);

    // 1) query f32 -> f16
    {
        int n = NROWS * DM;
        cvt_query_f16<<<(n + 255) / 256, 256, 0, stream>>>(query, Qh, n);
    }
    // 2) combined transposed f16 weight
    {
        int n = NCOLS * DM;
        build_wT_f16<<<(n + 255) / 256, 256, 0, stream>>>(W_off, W_attn, WhT);
    }
    // 3) value transposes: [bh][c][S] -> [bh][s][c]
    {
        dim3 blk(32, 8);
        transpose_vals<<<dim3((6400 + 31) / 32, 32), blk, 0, stream>>>(value0, vt0, 6400);
        transpose_vals<<<dim3((1600 + 31) / 32, 32), blk, 0, stream>>>(value1, vt1, 1600);
        transpose_vals<<<dim3(( 400 + 31) / 32, 32), blk, 0, stream>>>(value2, vt2, 400);
    }
    // 4) WMMA projection GEMM: 18 N-tiles x ceil(1350/8) M-blocks, 8 waves/block
    {
        dim3 grid(NCOLS / 16, (NROWS / 16 + 7) / 8);
        proj_wmma<<<grid, 256, 0, stream>>>(Qh, WhT, b_off, b_attn, P);
    }
    // 5) fused softmax + bilinear sample + weighted sum
    {
        msda_sample<<<NROWS, 256, 0, stream>>>(refpts, P, vt0, vt1, vt2, out);
    }
}
